// MultiInterestExtractor_47021301957248
// MI455X (gfx1250) — matrix operand order repaired and checked by reading, not compiled, not run on hardware
//
#include <hip/hip_runtime.h>
#include <hip/hip_bf16.h>

// MIND multi-interest routing for MI455X (gfx1250), wave32 + WMMA bf16.
// One workgroup (8 wave32) per batch element; 50 routing iterations run
// entirely out of LDS; all GEMMs use v_wmma_f32_16x16x32_bf16.

typedef __attribute__((ext_vector_type(16))) __bf16 v16bf;
typedef __attribute__((ext_vector_type(8)))  __bf16 v8bf;
typedef __attribute__((ext_vector_type(8)))  float  v8f;

#define S_DIM 200   // behaviors
#define SP    224   // S padded to multiple of 32 (bf16 WMMA K step)
#define DIN   64
#define DDIM  64
#define KINT  8     // interests
#define ITERS 50
#define BST   9     // b row stride in floats (odd -> conflict-free column reads)

__device__ __forceinline__ __bf16 f2bf(float f) {
  unsigned u = __builtin_bit_cast(unsigned, f);
  u += 0x7FFFu + ((u >> 16) & 1u);                 // round-to-nearest-even
  unsigned short s = (unsigned short)(u >> 16);
  return __builtin_bit_cast(__bf16, s);
}

__device__ __forceinline__ v16bf ld16(const __bf16* p0, const __bf16* p1) {
  // Two contiguous 16B LDS loads -> one WMMA operand (ds_load_b128 x2)
  union { v16bf v; v8bf h[2]; } u;
  u.h[0] = *(const v8bf*)p0;
  u.h[1] = *(const v8bf*)p1;
  return u.v;
}

__device__ __forceinline__ float wred_max(float v) {
  for (int o = 16; o; o >>= 1) v = fmaxf(v, __shfl_xor(v, o, 32));
  return v;
}
__device__ __forceinline__ float wred_sum(float v) {
  for (int o = 16; o; o >>= 1) v += __shfl_xor(v, o, 32);
  return v;
}

__global__ void __launch_bounds__(256)
mind_routing_wmma(const float* __restrict__ inputs,   // [B, 200, 64]
                  const float* __restrict__ W,        // [64, 64]
                  const float* __restrict__ bias,     // [64]
                  const float* __restrict__ b_init,   // [B, 200, 8]
                  float* __restrict__ out)            // [B, 8, 64]
{
  // LDS: v in two bf16 layouts (row-major for A-side, d-major for B-side),
  // wT (softmax weights, A-layout feed), u (B-layout feed), z + b in f32.
  __shared__ __align__(16) __bf16 sm_vT [DDIM * SP];   // v^T[d][s]; holds W^T during phase 0
  __shared__ __align__(16) __bf16 sm_vrm[SP * DDIM];   // v[s][d]; holds bf16(inputs) during phase 0
  __shared__ __align__(16) __bf16 sm_wT [16 * SP];     // w^T[k][s], rows 8..15 and s>=200 stay zero
  __shared__ __align__(16) __bf16 sm_urm[16 * DDIM];   // u[k][d], rows 8..15 stay zero
  __shared__ __align__(16) float  sm_z  [16 * DDIM];   // z (f32 accumulated)
  __shared__ __align__(16) float  sm_b  [SP * BST];    // routing logits (f32, stride 9)

  const int b    = blockIdx.x;
  const int tid  = threadIdx.x;
  const int lane = tid & 31;
  const int wv   = tid >> 5;       // wave id 0..7
  const int hlf  = lane >> 4;      // 0 | 1  (lane half, per WMMA layouts)
  const int lm   = lane & 15;
  const __bf16 bf0 = __builtin_bit_cast(__bf16, (unsigned short)0);

  // ---------------- Phase 0a: stage inputs(bf16) and W^T(bf16), b, zeros ----
  const float* inB = inputs + (size_t)b * S_DIM * DIN;
  for (int idx = tid; idx < SP * DIN; idx += 256)
    sm_vrm[idx] = (idx < S_DIM * DIN) ? f2bf(inB[idx]) : bf0;
  for (int idx = tid; idx < DIN * DDIM; idx += 256) {
    int d = idx >> 6, i = idx & 63;
    sm_vT[d * 64 + i] = f2bf(W[i * 64 + d]);         // W^T, stride 64 (phase-0 only)
  }
  for (int idx = tid; idx < 16 * SP; idx += 256) sm_wT[idx] = bf0;
  for (int idx = tid; idx < 8 * DDIM; idx += 256) sm_urm[8 * DDIM + idx] = bf0;
  for (int idx = tid; idx < SP * KINT; idx += 256) {
    int s = idx >> 3, k = idx & 7;
    sm_b[s * BST + k] = (s < S_DIM) ? b_init[(size_t)b * S_DIM * KINT + idx] : 0.0f;
  }
  __syncthreads();

  // ---------------- Phase 0b: v = inputs @ W + bias via WMMA ----------------
  // 13 S-tiles x 4 D-tiles = 52 tile jobs round-robin over 8 waves; results
  // buffered in accumulators, then written (bf16) into both v layouts.
  v8f acc0[7];
#pragma unroll
  for (int t = 0; t < 7; ++t) acc0[t] = (v8f){};
#pragma unroll
  for (int t = 0; t < 7; ++t) {
    int j = wv + 8 * t;
    if (j >= 52) break;
    int m = j >> 2, nt = j & 3;
    v8f acc = {};
#pragma unroll
    for (int c = 0; c < 2; ++c) {
      int srow = 16 * m + lm;
      int k0   = 32 * c + hlf * 8;
      v16bf A  = ld16(&sm_vrm[srow * 64 + k0], &sm_vrm[srow * 64 + k0 + 16]);
      int dcol = 16 * nt + lm;
      int i0   = 32 * c + hlf * 8;
      v16bf Bf = ld16(&sm_vT[dcol * 64 + i0], &sm_vT[dcol * 64 + i0 + 16]);
      acc = __builtin_amdgcn_wmma_f32_16x16x32_bf16(false, A, false, Bf,
                                                    (short)0, acc, false, false);
    }
    acc0[t] = acc;
  }
  __syncthreads();                                   // all W^T / input reads done
#pragma unroll
  for (int t = 0; t < 7; ++t) {
    int j = wv + 8 * t;
    if (j >= 52) break;
    int m = j >> 2, nt = j & 3;
#pragma unroll
    for (int jj = 0; jj < 8; ++jj) {
      int s = 16 * m + jj + hlf * 8;
      int d = 16 * nt + lm;
      if (s < S_DIM) {
        __bf16 h = f2bf(acc0[t][jj] + bias[d]);
        sm_vrm[s * 64 + d] = h;                      // row-major (GEMM2 A feed)
        sm_vT [d * SP + s] = h;                      // d-major   (GEMM1 B feed)
      }
    }
  }
  for (int idx = tid; idx < (SP - S_DIM) * DDIM; idx += 256) {
    int s = S_DIM + (idx >> 6), d = idx & 63;        // zero padded S rows
    sm_vrm[s * 64 + d] = bf0;
    sm_vT [d * SP + s] = bf0;
  }
  __syncthreads();

  // ---------------- Routing loop (all in LDS) -------------------------------
  for (int it = 0; it < ITERS; ++it) {
    const bool is_final = (it == ITERS - 1);

    // softmax over s for interest k = wv (wave per column; stride-9 reads)
    {
      int k = wv;
      float vals[7];
      float mx = -3.4e38f;
#pragma unroll
      for (int t = 0; t < 7; ++t) {
        int s = lane + 32 * t;
        float bb = (s < S_DIM) ? sm_b[s * BST + k] : -3.4e38f;
        vals[t] = bb;
        mx = fmaxf(mx, bb);
      }
      mx = wred_max(mx);
      float sum = 0.0f;
#pragma unroll
      for (int t = 0; t < 7; ++t) {
        int s = lane + 32 * t;
        float e = (s < S_DIM) ? __expf(vals[t] - mx) : 0.0f;
        vals[t] = e;
        sum += e;
      }
      float inv = 1.0f / wred_sum(sum);
#pragma unroll
      for (int t = 0; t < 7; ++t) {
        int s = lane + 32 * t;
        if (s < S_DIM) sm_wT[k * SP + s] = f2bf(vals[t] * inv);
      }
    }
    __syncthreads();

    // GEMM1: z[16x64] = w^T(A) x v(B); waves 0..3, 7 K-chunks of 32 over S
    if (wv < 4) {
      v8f acc = {};
#pragma unroll
      for (int c = 0; c < 7; ++c) {
        int k0   = 32 * c + hlf * 8;
        v16bf A  = ld16(&sm_wT[lm * SP + k0], &sm_wT[lm * SP + k0 + 16]);
        int d    = 16 * wv + lm;
        int s0   = 32 * c + hlf * 8;
        v16bf Bf = ld16(&sm_vT[d * SP + s0], &sm_vT[d * SP + s0 + 16]);
        acc = __builtin_amdgcn_wmma_f32_16x16x32_bf16(false, A, false, Bf,
                                                      (short)0, acc, false, false);
      }
#pragma unroll
      for (int jj = 0; jj < 8; ++jj) {
        int M = jj + hlf * 8;
        sm_z[M * 64 + 16 * wv + lm] = acc[jj];
      }
    }
    __syncthreads();

    // squash row k = wv (f32); final iteration writes output directly
    {
      int k = wv;
      float z0 = sm_z[k * 64 + lane];
      float z1 = sm_z[k * 64 + lane + 32];
      float ss = wred_sum(z0 * z0 + z1 * z1);
      float ns = ss * (1.0f / 64.0f);
      float sc = ns / ((1.0f + ns) * sqrtf(ns + 1e-9f));
      if (is_final) {
        out[(size_t)b * KINT * DDIM + k * DDIM + lane]      = sc * z0;
        out[(size_t)b * KINT * DDIM + k * DDIM + lane + 32] = sc * z1;
      } else {
        sm_urm[k * 64 + lane]      = f2bf(sc * z0);
        sm_urm[k * 64 + lane + 32] = f2bf(sc * z1);
      }
    }

    if (!is_final) {
      __syncthreads();
      // GEMM2: b[s,k] += v(A) x u^T(B); 13 S-tiles over 8 waves, K=64 (2 chunks)
#pragma unroll
      for (int rep = 0; rep < 2; ++rep) {
        int m = wv + 8 * rep;
        if (m < 13) {                               // wave-uniform: EXEC all-1 for WMMA
          v8f acc = {};
#pragma unroll
          for (int c = 0; c < 2; ++c) {
            int srow = 16 * m + lm;
            int k0   = 32 * c + hlf * 8;
            v16bf A  = ld16(&sm_vrm[srow * 64 + k0], &sm_vrm[srow * 64 + k0 + 16]);
            int d0   = 32 * c + hlf * 8;
            v16bf Bf = ld16(&sm_urm[lm * 64 + d0], &sm_urm[lm * 64 + d0 + 16]);
            acc = __builtin_amdgcn_wmma_f32_16x16x32_bf16(false, A, false, Bf,
                                                          (short)0, acc, false, false);
          }
#pragma unroll
          for (int jj = 0; jj < 8; ++jj) {
            int s = 16 * m + jj + hlf * 8;
            if (lm < KINT) sm_b[s * BST + lm] += acc[jj];
          }
        }
      }
      __syncthreads();
    }
  }
}

extern "C" void kernel_launch(void* const* d_in, const int* in_sizes, int n_in,
                              void* d_out, int out_size, void* d_ws, size_t ws_size,
                              hipStream_t stream) {
  (void)n_in; (void)d_ws; (void)ws_size; (void)out_size;
  const float* inputs = (const float*)d_in[0];
  const float* W      = (const float*)d_in[1];
  const float* bias   = (const float*)d_in[2];
  const float* b_init = (const float*)d_in[3];
  float* out          = (float*)d_out;
  const int B = in_sizes[0] / (S_DIM * DIN);   // 2048
  mind_routing_wmma<<<B, 256, 0, stream>>>(inputs, W, bias, b_init, out);
}
